// Prototype_35940286332970
// MI455X (gfx1250) — matrix operand order repaired and checked by reading, not compiled, not run on hardware
//
#include <hip/hip_runtime.h>

typedef _Float16 v16h __attribute__((ext_vector_type(16)));
typedef float    v8f  __attribute__((ext_vector_type(8)));

#define CEILDIV(a, b) (((a) + (b) - 1) / (b))
static const int BB = 8;          // batch
static const int NN = 16384;      // input length
static const float EPSF = 1e-5f;
static const float SLOPEF = 0.01f;

// ---- gfx1250 async global->LDS (probe-verified: takes int* params) ----
#if defined(__has_builtin)
#if __has_builtin(__builtin_amdgcn_global_load_async_to_lds_b32) && \
    __has_builtin(__builtin_amdgcn_s_wait_asynccnt)
#define ASYNC_LDS 1
#endif
#endif
#define AS1P(p) ((__attribute__((address_space(1))) int*)(p))
#define AS3P(p) ((__attribute__((address_space(3))) int*)(p))

// ---------------------------------------------------------------------------
// enc0: Cin=1, k=3, stride=2 conv. x: (B, N, 1) == (B, 1, N). grid: (L/256, 64, B)
// ---------------------------------------------------------------------------
__global__ void enc0_conv(const float* __restrict__ x, const float* __restrict__ w,
                          const float* __restrict__ bias, float* __restrict__ out,
                          int Lin, int Lout) {
  int l = blockIdx.x * blockDim.x + threadIdx.x;
  if (l >= Lout) return;
  int co = blockIdx.y;
  int b  = blockIdx.z;
  const float* xb = x + (size_t)b * Lin;
  float s = bias[co];
#pragma unroll
  for (int t = 0; t < 3; ++t) {
    int pos = 2 * l - 1 + t;
    if (pos >= 0 && pos < Lin) s += w[co * 3 + t] * xb[pos];
  }
  out[((size_t)b * gridDim.y + co) * Lout + l] = s;
}

// ---------------------------------------------------------------------------
// BN training stats: per-channel sum & sumsq over (B, L). One block / channel.
// ---------------------------------------------------------------------------
__global__ __launch_bounds__(256)
void bn_stats(const float* __restrict__ x, float* __restrict__ stats,
              int Bn, int C, int L) {
  int c = blockIdx.x;
  float s = 0.f, s2 = 0.f;
  for (int b = 0; b < Bn; ++b) {
    const float* xb = x + ((size_t)b * C + c) * L;
    for (int l = threadIdx.x; l < L; l += 256) {
      float v = xb[l];
      s += v;
      s2 += v * v;
    }
  }
  __shared__ float sh[256];
  __shared__ float sh2[256];
  sh[threadIdx.x] = s;
  sh2[threadIdx.x] = s2;
  __syncthreads();
  for (int o = 128; o > 0; o >>= 1) {
    if ((int)threadIdx.x < o) {
      sh[threadIdx.x]  += sh[threadIdx.x + o];
      sh2[threadIdx.x] += sh2[threadIdx.x + o];
    }
    __syncthreads();
  }
  if (threadIdx.x == 0) {
    stats[c]     = sh[0];
    stats[C + c] = sh2[0];
  }
}

// ---------------------------------------------------------------------------
// Apply BN (batch stats) + LeakyReLU in place. grid: (L/256, C, B)
// ---------------------------------------------------------------------------
__global__ void bn_apply(float* __restrict__ x, const float* __restrict__ stats,
                         const float* __restrict__ g, const float* __restrict__ be,
                         int Bn, int C, int L) {
  int l = blockIdx.x * blockDim.x + threadIdx.x;
  if (l >= L) return;
  int c = blockIdx.y;
  int b = blockIdx.z;
  size_t idx = ((size_t)b * C + c) * L + l;
  float inv = 1.f / (float)(Bn * L);
  float m   = stats[c] * inv;
  float var = stats[C + c] * inv - m * m;
  float sc  = g[c] * rsqrtf(var + EPSF);
  float v = sc * (x[idx] - m) + be[c];
  x[idx] = (v >= 0.f) ? v : SLOPEF * v;
}

// ---------------------------------------------------------------------------
// Pack conv weights (Cout, K=Cin*ks) into WMMA A-fragment order (f16):
// apack[((mt*nsteps + s)*32 + lane)*16 + e]  == afrag element e of `lane` for
// M-tile mt at K-step s (contiguous k order: k = s*32 + 16*(lane/16) + e).
// ---------------------------------------------------------------------------
__global__ void pack_weights(const float* __restrict__ w, _Float16* __restrict__ apack,
                             int Cout, int K, int nsteps, int total) {
  int idx = blockIdx.x * blockDim.x + threadIdx.x;
  if (idx >= total) return;
  int e    = idx & 15;
  int lane = (idx >> 4) & 31;
  int rest = idx >> 9;
  int s    = rest % nsteps;
  int mt   = rest / nsteps;
  int lr = lane & 15, hx = lane >> 4;
  int row = mt * 16 + lr;
  int j = e >> 1, r = e & 1;
  int ka = s * 32 + 16 * hx + 2 * j + r;
  float v = (row < Cout && ka < K) ? w[(size_t)row * K + ka] : 0.f;
  apack[idx] = (_Float16)v;
}

// ---------------------------------------------------------------------------
// Implicit-GEMM conv, WMMA f32<=f16 16x16x32.
// Block = 256 threads = 8 waves = 8 M-tiles sharing one LDS input patch.
// Each wave holds CT accumulators (CT x 16 columns) and reuses each A fragment
// for CT back-to-back v_wmma ops. B fragments are gathered from the LDS patch
// (f32, converted at use). Staging uses gfx1250 async LDS DMA (ASYNCcnt).
// No EXEC divergence: all bounds handled with predicated loads.
// ---------------------------------------------------------------------------
template <int KS, int STRIDE, int CT>
__global__ __launch_bounds__(256)
void conv_wmma2(const float* __restrict__ in, const _Float16* __restrict__ apack,
                const float* __restrict__ bias, float* __restrict__ out,
                int Cin, int Lin, int Cout, int Lout,
                int mGroups, int nsteps, int K) {
  constexpr int COLS = 16 * CT;
  constexpr int SPAN = COLS * STRIDE + KS - 1;
  constexpr int PAD  = (KS - 1) / 2;
  extern __shared__ char smem_raw[];
  float* patch = (float*)smem_raw;  // [Cin][SPAN]

  const int bm = blockIdx.x % mGroups;
  const int bn = blockIdx.x / mGroups;
  const int nbase = bn * COLS;
  const int b     = nbase / Lout;          // COLS | Lout => no batch straddle
  const int lbase = nbase - b * Lout;
  const float* inb = in + (size_t)b * Cin * Lin;

  // ---- stage input patch into LDS ----
  const int totalStage = Cin * SPAN;
  const int basePos = lbase * STRIDE - PAD;
  for (int i = threadIdx.x; i < totalStage; i += 256) {
    int ci = i / SPAN;                     // SPAN is constexpr -> cheap
    int x  = i - ci * SPAN;
    int pos = basePos + x;
#ifdef ASYNC_LDS
    if (pos >= 0 && pos < Lin) {
      __builtin_amdgcn_global_load_async_to_lds_b32(
          AS1P(inb + (size_t)ci * Lin + pos), AS3P(&patch[i]), 0, 0);
    } else {
      patch[i] = 0.f;
    }
#else
    float v = 0.f;
    if (pos >= 0 && pos < Lin) v = inb[(size_t)ci * Lin + pos];
    patch[i] = v;
#endif
  }
#ifdef ASYNC_LDS
  __builtin_amdgcn_s_wait_asynccnt(0);
#endif
  __syncthreads();

  // ---- WMMA main loop ----
  const int lane = threadIdx.x & 31;
  const int wv   = threadIdx.x >> 5;
  const int lr = lane & 15;
  const int hx = lane >> 4;
  const int mt = bm * 8 + wv;
  const int m0 = mt * 16;

  v8f acc[CT];
#pragma unroll
  for (int ct = 0; ct < CT; ++ct) acc[ct] = (v8f){0.f, 0.f, 0.f, 0.f, 0.f, 0.f, 0.f, 0.f};

  const v16h* ap = (const v16h*)apack + (size_t)mt * nsteps * 32 + lane;
  for (int s = 0; s < nsteps; ++s, ap += 32) {
    v16h afrag = *ap;                      // prepacked A fragment (2x16B loads)
    const int kbase = s * 32 + 16 * hx;
#pragma unroll
    for (int ct = 0; ct < CT; ++ct) {
      v16h bfrag;
#pragma unroll
      for (int e = 0; e < 16; ++e) {
        int k  = kbase + e;
        int ci = k / KS;                   // KS constexpr
        int t  = k - ci * KS;
        int x  = (ct * 16 + lr) * STRIDE + t;
        float v = (k < K) ? patch[ci * SPAN + x] : 0.f;
        bfrag[e] = (_Float16)v;
      }
      acc[ct] = __builtin_amdgcn_wmma_f32_16x16x32_f16(
          false, afrag, false, bfrag, (short)0, acc[ct], false, false);
    }
  }

  // ---- epilogue: bias + store (rows always valid: tilesM covers Cout) ----
  float* outb = out + (size_t)b * Cout * Lout;
#pragma unroll
  for (int i = 0; i < 8; ++i) {
    int row = m0 + i + 8 * hx;
    float br = bias[row];
#pragma unroll
    for (int ct = 0; ct < CT; ++ct) {
      outb[(size_t)row * Lout + (lbase + ct * 16 + lr)] = acc[ct][i] + br;
    }
  }
}

// ---------------------------------------------------------------------------
// Banded correlation, K=3 clamped patches, D=5. grid: (N/256, 11, B)
// ---------------------------------------------------------------------------
__global__ void fusion_corr(const float* __restrict__ f, float* __restrict__ feat,
                            int C, int N) {
  int i = blockIdx.x * blockDim.x + threadIdx.x;
  if (i >= N) return;
  int dd = blockIdx.y;
  int b  = blockIdx.z;
  int d = dd - 5;
  int j = min(max(i + d, 0), N - 1);
  const float* fb = f + (size_t)b * C * N;
  int a0 = max(i - 1, 0), a1 = i, a2 = min(i + 1, N - 1);
  int c0 = max(j - 1, 0), c1 = j, c2 = min(j + 1, N - 1);
  float s = 0.f;
  for (int c = 0; c < C; ++c) {
    const float* row = fb + (size_t)c * N;
    s += row[a0] * row[c0] + row[a1] * row[c1] + row[a2] * row[c2];
  }
  feat[((size_t)b * 11 + dd) * N + i] = s;
}

// ---------------------------------------------------------------------------
// Nearest x2 upsample of src2 + concat [src1 (C1,Lout) ; src2 (C2,Lout/2)].
// grid: (Lout/256, C1+C2, B)
// ---------------------------------------------------------------------------
__global__ void upsample_concat(const float* __restrict__ s1, const float* __restrict__ s2,
                                float* __restrict__ out, int C1, int C2, int Lout) {
  int l = blockIdx.x * blockDim.x + threadIdx.x;
  if (l >= Lout) return;
  int c = blockIdx.y;
  int b = blockIdx.z;
  int Ct = C1 + C2;
  float v;
  if (c < C1) v = s1[((size_t)b * C1 + c) * Lout + l];
  else        v = s2[((size_t)b * C2 + (c - C1)) * (Lout >> 1) + (l >> 1)];
  out[((size_t)b * Ct + c) * Lout + l] = v;
}

// ---------------------------------------------------------------------------
// 1x1 flow head conv: (B, 129, L) -> raw (B, 2, L). grid: (L/256, 2, B)
// ---------------------------------------------------------------------------
__global__ void flow_conv(const float* __restrict__ in, const float* __restrict__ w,
                          const float* __restrict__ bias, float* __restrict__ out,
                          int Cin, int L) {
  int l = blockIdx.x * blockDim.x + threadIdx.x;
  if (l >= L) return;
  int co = blockIdx.y;
  int b  = blockIdx.z;
  const float* ib = in + (size_t)b * Cin * L + l;
  float s = bias[co];
  for (int ci = 0; ci < Cin; ++ci) s += w[co * Cin + ci] * ib[(size_t)ci * L];
  out[((size_t)b * 2 + co) * L + l] = s;
}

// ---------------------------------------------------------------------------
// Final BN + LReLU + transpose to (B, L, 2). grid: (L/256, 2, B)
// ---------------------------------------------------------------------------
__global__ void flow_finalize(const float* __restrict__ raw, const float* __restrict__ stats,
                              const float* __restrict__ g, const float* __restrict__ be,
                              float* __restrict__ out, int Bn, int L) {
  int l = blockIdx.x * blockDim.x + threadIdx.x;
  if (l >= L) return;
  int co = blockIdx.y;
  int b  = blockIdx.z;
  float inv = 1.f / (float)(Bn * L);
  float m   = stats[co] * inv;
  float var = stats[2 + co] * inv - m * m;
  float sc  = g[co] * rsqrtf(var + EPSF);
  float v = sc * (raw[((size_t)b * 2 + co) * L + l] - m) + be[co];
  v = (v >= 0.f) ? v : SLOPEF * v;
  out[((size_t)b * L + l) * 2 + co] = v;
}

// ---------------------------------------------------------------------------
static void launch_bn(float* x, float* stats, const float* g, const float* be,
                      int Bn, int C, int L, hipStream_t s) {
  bn_stats<<<C, 256, 0, s>>>(x, stats, Bn, C, L);
  bn_apply<<<dim3(CEILDIV(L, 256), C, Bn), 256, 0, s>>>(x, stats, g, be, Bn, C, L);
}

extern "C" void kernel_launch(void* const* d_in, const int* in_sizes, int n_in,
                              void* d_out, int out_size, void* d_ws, size_t ws_size,
                              hipStream_t stream) {
  const float* scan1 = (const float*)d_in[0];
  const float *enc0_w = (const float*)d_in[1],  *enc0_b = (const float*)d_in[2],
              *enc0_g = (const float*)d_in[3],  *enc0_be = (const float*)d_in[4];
  const float *enc1_w = (const float*)d_in[5],  *enc1_b = (const float*)d_in[6],
              *enc1_g = (const float*)d_in[7],  *enc1_be = (const float*)d_in[8];
  const float *enc2_w = (const float*)d_in[9],  *enc2_b = (const float*)d_in[10],
              *enc2_g = (const float*)d_in[11], *enc2_be = (const float*)d_in[12];
  const float *dec1_w = (const float*)d_in[13], *dec1_b = (const float*)d_in[14],
              *dec1_g = (const float*)d_in[15], *dec1_be = (const float*)d_in[16];
  const float *dec0_w = (const float*)d_in[17], *dec0_b = (const float*)d_in[18],
              *dec0_g = (const float*)d_in[19], *dec0_be = (const float*)d_in[20];
  const float *flow_w = (const float*)d_in[21], *flow_b = (const float*)d_in[22],
              *flow_g = (const float*)d_in[23], *flow_be = (const float*)d_in[24];

  float* ws = (float*)d_ws;
  const int L0 = 8192, L1 = 4096, L2 = 2048;

  // Aliased workspace (float offsets); ~121.2 MB peak.
  float* f0      = ws + 0;          // 8*64*8192  [0, 4194304)
  float* f1      = ws + 4194304;    // 8*128*4096 [4194304, 8388608)
  float* f2      = ws + 8388608;    // 8*256*2048 [8388608, 12582912)
  float* feat    = ws + 12582912;   // 8*11*2048  [12582912, 12763136)
  float* cat1    = ws + 12763136;   // 8*139*4096 [12763136, 17317888)
  float* cat0    = ws + 17317888;   // 8*192*8192 [17317888, 29900800)
  float* dec1out = ws + 8388608;    // reuse f2 (dead after fusion)
  float* dec0out = ws + 0;          // reuse f0+f1 (dead after cat0/cat1)
  float* catF    = ws + 8388608;    // reuse f2/feat/cat1/dec1 + low cat0
  float* flowraw = ws + 29900800;   // 8*2*16384
  float* stats   = ws + 30162944;   // 512 floats
  _Float16* apackBase = (_Float16*)(ws + 30163456);
  _Float16* ap_enc1 = apackBase + 0;        // 8*6*512   = 24576 halves
  _Float16* ap_enc2 = apackBase + 24576;    // 16*12*512 = 98304
  _Float16* ap_dec1 = apackBase + 122880;   // 8*14*512  = 57344
  _Float16* ap_dec0 = apackBase + 180224;   // 8*18*512  = 73728

  // ---- pack all conv weights into WMMA fragment order ----
  pack_weights<<<CEILDIV(24576, 256), 256, 0, stream>>>(enc1_w, ap_enc1, 128, 192, 6,  24576);
  pack_weights<<<CEILDIV(98304, 256), 256, 0, stream>>>(enc2_w, ap_enc2, 256, 384, 12, 98304);
  pack_weights<<<CEILDIV(57344, 256), 256, 0, stream>>>(dec1_w, ap_dec1, 128, 417, 14, 57344);
  pack_weights<<<CEILDIV(73728, 256), 256, 0, stream>>>(dec0_w, ap_dec0, 128, 576, 18, 73728);

  // ---- encoder ----
  enc0_conv<<<dim3(CEILDIV(L0, 256), 64, BB), 256, 0, stream>>>(scan1, enc0_w, enc0_b, f0, NN, L0);
  launch_bn(f0, stats, enc0_g, enc0_be, BB, 64, L0, stream);

  // enc1: 64->128, stride 2. grid = mGroups(1) * nGroups(32768/64)
  conv_wmma2<3, 2, 4><<<512, 256, 64 * 130 * 4, stream>>>(
      f0, ap_enc1, enc1_b, f1, 64, L0, 128, L1, 1, 6, 192);
  launch_bn(f1, stats, enc1_g, enc1_be, BB, 128, L1, stream);

  // enc2: 128->256, stride 2. mGroups=2, nGroups=16384/32=512
  conv_wmma2<3, 2, 2><<<1024, 256, 128 * 66 * 4, stream>>>(
      f1, ap_enc2, enc2_b, f2, 128, L1, 256, L2, 2, 12, 384);
  launch_bn(f2, stats, enc2_g, enc2_be, BB, 256, L2, stream);

  // ---- correlation (scan2 == scan1 -> f2 vs f2) ----
  fusion_corr<<<dim3(CEILDIV(L2, 256), 11, BB), 256, 0, stream>>>(f2, feat, 256, L2);

  // ---- decoder level 1 ----
  upsample_concat<<<dim3(CEILDIV(L1, 256), 139, BB), 256, 0, stream>>>(f1, feat, cat1, 128, 11, L1);
  conv_wmma2<3, 1, 4><<<512, 256, 139 * 66 * 4, stream>>>(
      cat1, ap_dec1, dec1_b, dec1out, 139, L1, 128, L1, 1, 14, 417);
  launch_bn(dec1out, stats, dec1_g, dec1_be, BB, 128, L1, stream);

  // ---- decoder level 0 ----
  upsample_concat<<<dim3(CEILDIV(L0, 256), 192, BB), 256, 0, stream>>>(f0, dec1out, cat0, 64, 128, L0);
  conv_wmma2<3, 1, 4><<<1024, 256, 192 * 66 * 4, stream>>>(
      cat0, ap_dec0, dec0_b, dec0out, 192, L0, 128, L0, 1, 18, 576);
  launch_bn(dec0out, stats, dec0_g, dec0_be, BB, 128, L0, stream);

  // ---- flow head ----
  upsample_concat<<<dim3(CEILDIV(NN, 256), 129, BB), 256, 0, stream>>>(scan1, dec0out, catF, 1, 128, NN);
  flow_conv<<<dim3(CEILDIV(NN, 256), 2, BB), 256, 0, stream>>>(catF, flow_w, flow_b, flowraw, 129, NN);
  bn_stats<<<2, 256, 0, stream>>>(flowraw, stats, BB, 2, NN);
  flow_finalize<<<dim3(CEILDIV(NN, 256), 2, BB), 256, 0, stream>>>(flowraw, stats, flow_g, flow_be,
                                                                   (float*)d_out, BB, NN);
}
